// MinEuclideanDistBlock_2826088480903
// MI455X (gfx1250) — matrix operand order repaired and checked by reading, compiled, not run on hardware
//
#include <hip/hip_runtime.h>
#include <hip/hip_bf16.h>
#include <math.h>

// Problem constants (match reference)
#define B_ 32
#define C_ 8
#define L_ 4096
#define K_ 128
#define S_ 128
#define W_ (L_ - S_ + 1)   // 3969
#define NWB 64             // window positions per workgroup

typedef __attribute__((ext_vector_type(16))) __bf16 v16bf;
typedef __attribute__((ext_vector_type(8)))  float  v8f;

// ---- workspace layout (bytes) ----
#define OFF_XH  0                                   // x hi bf16: B*C*L
#define OFF_XL  (OFF_XH + 2*B_*C_*L_)               // x lo bf16
#define OFF_SH  (OFF_XL + 2*B_*C_*L_)               // shapelets hi bf16: C*K*S
#define OFF_SL  (OFF_SH + 2*C_*K_*S_)               // shapelets lo bf16
#define OFF_SN  (OFF_SL + 2*C_*K_*S_)               // snorm f32: C*K
#define OFF_XN  (OFF_SN + 4*C_*K_)                  // xnorm f32: B*C*W
// total ~8.8 MB

#define FLTMAX 3.402823466e+38f

// ---------------- pre-pass kernels ----------------

__global__ void init_out_kernel(float* out) {
  int i = blockIdx.x * blockDim.x + threadIdx.x;
  if (i < B_ * K_) out[i] = FLTMAX;
}

__global__ void convert_x_kernel(const float* __restrict__ x,
                                 __bf16* __restrict__ xh, __bf16* __restrict__ xl) {
  int i = blockIdx.x * blockDim.x + threadIdx.x;
  if (i < B_ * C_ * L_) {
    float v = x[i];
    __bf16 h = (__bf16)v;
    xh[i] = h;
    xl[i] = (__bf16)(v - (float)h);
  }
}

__global__ void convert_s_kernel(const float* __restrict__ s,
                                 __bf16* __restrict__ sh, __bf16* __restrict__ sl,
                                 float* __restrict__ sn) {
  int r = blockIdx.x * blockDim.x + threadIdx.x;   // r = c*K + k
  if (r < C_ * K_) {
    const float* row = s + r * S_;
    float acc = 0.f;
    #pragma unroll 8
    for (int i = 0; i < S_; ++i) {
      float v = row[i];
      acc += v * v;
      __bf16 h = (__bf16)v;
      sh[r * S_ + i] = h;
      sl[r * S_ + i] = (__bf16)(v - (float)h);
    }
    sn[r] = acc;
  }
}

__global__ void xnorm_kernel(const float* __restrict__ x, float* __restrict__ xn) {
  int i = blockIdx.x * blockDim.x + threadIdx.x;   // over B*C*W
  if (i < B_ * C_ * W_) {
    int row = i / W_;                  // row = b*C + c (256 rows)
    int w   = i - row * W_;
    const float* p = x + row * L_ + w;
    float acc = 0.f;
    #pragma unroll 8
    for (int s = 0; s < S_; ++s) { float v = p[s]; acc += v * v; }
    xn[i] = acc;
  }
}

// ---------------- fragment loaders (contiguous k order k = 16*half + s, not the ISA table order) ----------------

// B matrix 32x16 (K x N), 16-bit: lanes 0-15 hold K=0..15, lanes 16-31 hold
// K=16..31; column N = lane%16; per-lane the 16 halves are contiguous in s,
// which for a stride-1 window is contiguous memory.
__device__ inline v16bf load_bfrag(const __bf16* p) {
  union { v16bf v; __bf16 e[16]; } u;
  #pragma unroll
  for (int i = 0; i < 16; ++i) u.e[i] = p[i];
  return u.v;
}

// A matrix 16x32 (M x K), 16-bit: lane holds row M = lane%16; lanes own
// K = s0..s0+15 with s0 = sc*32 (+16 for lanes>=16).
__device__ inline v16bf load_afrag(const __bf16* row, int s0) {
  union { v16bf v; __bf16 e[16]; } u;
  #pragma unroll
  for (int i = 0; i < 8; ++i) {
    u.e[i]     = row[s0 + i];
    u.e[8 + i] = row[s0 + 8 + i];
  }
  return u.v;
}

// ---------------- main WMMA kernel ----------------

__global__ __launch_bounds__(256)
void shapelet_min_kernel(const __bf16* __restrict__ xh, const __bf16* __restrict__ xl,
                         const __bf16* __restrict__ sh, const __bf16* __restrict__ sl,
                         const float* __restrict__ sn, const float* __restrict__ xn,
                         unsigned int* __restrict__ out) {
  __shared__ __bf16 xwin_h[NWB + S_];     // 192 (need NWB+S-1 = 191)
  __shared__ __bf16 xwin_l[NWB + S_];
  __shared__ float  xnorm_sh[NWB];
  __shared__ float  snorm_sh[K_];

  const int wblock = blockIdx.x;
  const int b      = blockIdx.y;
  const int w0     = wblock * NWB;
  const int tid    = threadIdx.x;
  const int wave   = tid >> 5;            // wave32
  const int lane   = tid & 31;
  const int ln16   = lane & 15;
  const int hb     = (lane >> 4) & 1;

  const int kbase = wave * 16;            // this wave's 16 shapelets
  const int krow  = kbase + ln16;         // A-fragment row for this lane

  float dsum[4][8];                       // sum over channels of sqrt(d2)
  #pragma unroll
  for (int t = 0; t < 4; ++t)
    #pragma unroll
    for (int v = 0; v < 8; ++v) dsum[t][v] = 0.f;

  for (int c = 0; c < C_; ++c) {
    // ---- cooperative stage of this channel's window + norms into LDS ----
    if (tid < NWB + S_ - 1) {
      int pos = w0 + tid;
      __bf16 z = (__bf16)0.f;
      xwin_h[tid] = (pos < L_) ? xh[(b * C_ + c) * L_ + pos] : z;
      xwin_l[tid] = (pos < L_) ? xl[(b * C_ + c) * L_ + pos] : z;
    }
    if (tid < NWB) {
      int w = w0 + tid;
      xnorm_sh[tid] = (w < W_) ? xn[(b * C_ + c) * W_ + w] : 0.f;
    }
    if (tid >= 64 && tid < 64 + K_) {
      snorm_sh[tid - 64] = sn[c * K_ + (tid - 64)];
    }
    __syncthreads();

    // ---- A fragments (shapelets, bf16 hi/lo) for this wave's k-tile ----
    const __bf16* rowh = sh + (c * K_ + krow) * S_;
    const __bf16* rowl = sl + (c * K_ + krow) * S_;
    v16bf Ah[4], Al[4];
    #pragma unroll
    for (int sc = 0; sc < 4; ++sc) {
      int s0 = sc * 32 + hb * 16;
      Ah[sc] = load_afrag(rowh, s0);
      Al[sc] = load_afrag(rowl, s0);
    }

    // ---- 4 window tiles of 16 positions each ----
    #pragma unroll
    for (int t = 0; t < 4; ++t) {
      v8f acc = {0.f, 0.f, 0.f, 0.f, 0.f, 0.f, 0.f, 0.f};
      #pragma unroll
      for (int sc = 0; sc < 4; ++sc) {
        int base = t * 16 + ln16 + sc * 32 + hb * 16;   // max 190 < 192
        v16bf Bh = load_bfrag(&xwin_h[base]);
        v16bf Bl = load_bfrag(&xwin_l[base]);
        // bf16x3 split: hi*hi + hi*lo + lo*hi, f32 accumulate (~f32 accuracy)
        acc = __builtin_amdgcn_wmma_f32_16x16x32_bf16(false, Ah[sc], false, Bh,
                                                      (short)0, acc, false, false);
        acc = __builtin_amdgcn_wmma_f32_16x16x32_bf16(false, Ah[sc], false, Bl,
                                                      (short)0, acc, false, false);
        acc = __builtin_amdgcn_wmma_f32_16x16x32_bf16(false, Al[sc], false, Bh,
                                                      (short)0, acc, false, false);
      }
      // epilogue: d2 = ||x_w||^2 + ||s_k||^2 - 2*cross ; accumulate sqrt
      // C/D layout: element v of lane -> M = v + 8*hb, N = ln16
      float xnv = xnorm_sh[t * 16 + ln16];
      #pragma unroll
      for (int v = 0; v < 8; ++v) {
        float d2 = xnv + snorm_sh[kbase + v + hb * 8] - 2.0f * acc[v];
        dsum[t][v] += sqrtf(fmaxf(d2, 1e-12f));
      }
    }
    __syncthreads();
  }

  // ---- min over window positions, then global atomic min ----
  #pragma unroll
  for (int v = 0; v < 8; ++v) {
    float m = FLTMAX;
    #pragma unroll
    for (int t = 0; t < 4; ++t) {
      int w = w0 + t * 16 + ln16;
      float val = (w < W_) ? dsum[t][v] : FLTMAX;
      m = fminf(m, val);
    }
    // min across the 16 lanes of each half-wave (w varies with ln16; k fixed)
    #pragma unroll
    for (int mask = 1; mask < 16; mask <<= 1)
      m = fminf(m, __shfl_xor(m, mask, 32));
    if (ln16 == 0) {
      int k = kbase + v + hb * 8;
      // distances are >= 0, so uint compare == float compare
      atomicMin(&out[b * K_ + k], __float_as_uint(m));
    }
  }
}

// ---------------- launcher ----------------

extern "C" void kernel_launch(void* const* d_in, const int* in_sizes, int n_in,
                              void* d_out, int out_size, void* d_ws, size_t ws_size,
                              hipStream_t stream) {
  (void)in_sizes; (void)n_in; (void)out_size; (void)ws_size;
  const float* x   = (const float*)d_in[0];   // (B, C, L) f32
  const float* shp = (const float*)d_in[1];   // (C, K, S) f32

  char* ws = (char*)d_ws;
  __bf16* xh = (__bf16*)(ws + OFF_XH);
  __bf16* xl = (__bf16*)(ws + OFF_XL);
  __bf16* sH = (__bf16*)(ws + OFF_SH);
  __bf16* sL = (__bf16*)(ws + OFF_SL);
  float*  sn = (float*)(ws + OFF_SN);
  float*  xn = (float*)(ws + OFF_XN);

  init_out_kernel<<<(B_ * K_ + 255) / 256, 256, 0, stream>>>((float*)d_out);
  convert_x_kernel<<<(B_ * C_ * L_ + 255) / 256, 256, 0, stream>>>(x, xh, xl);
  convert_s_kernel<<<(C_ * K_ + 255) / 256, 256, 0, stream>>>(shp, sH, sL, sn);
  xnorm_kernel<<<(B_ * C_ * W_ + 255) / 256, 256, 0, stream>>>(x, xn);

  dim3 grid((W_ + NWB - 1) / NWB, B_);    // (63, 32)
  shapelet_min_kernel<<<grid, 256, 0, stream>>>(xh, xl, sH, sL, sn, xn,
                                                (unsigned int*)d_out);
}
